// Attention_40948218200647
// MI455X (gfx1250) — matrix operand order repaired and checked by reading, not compiled, not run on hardware
//
#include <hip/hip_runtime.h>

typedef __attribute__((ext_vector_type(16))) __bf16 v16bf;
typedef __attribute__((ext_vector_type(8)))  __bf16 v8bf;
typedef __attribute__((ext_vector_type(4)))  __bf16 v4bf;
typedef __attribute__((ext_vector_type(8)))  float  v8f;

#define LDT 40   // staging LDS row stride (bf16); 80B rows -> conflict-free b128 frag loads
#define LDE 136  // epilogue transpose-tile row stride (bf16)

#define ASZ (64 * LDT)        // one A staging buffer
#define BSZ (128 * LDT)       // one B staging buffer
#define BOFF (2 * ASZ)        // B buffers start after the two A buffers
#define SMEM_ELEMS (2 * ASZ + 2 * BSZ)  // 15360 bf16 = 30720 B; epilogue tile (64*LDE=8704) reuses it

#define WMMA_BF16(a, b, c) \
  __builtin_amdgcn_wmma_f32_16x16x32_bf16(false, (a), false, (b), (short)0, (c), false, false)

// ---- staging: ROWS x 32 tile into LDS, 128 threads ---------------------------

template <int ROWS>
__device__ __forceinline__ void stage_f32(const float* __restrict__ src, int ld,
                                          __bf16* dst, int tid) {
  constexpr int IT = ROWS / 16;  // per iter: 128 thr * 4 floats = 16 rows
  float4 f[IT];
#pragma unroll
  for (int it = 0; it < IT; ++it) {
    int e = (it * 128 + tid) * 4;
    int r = e >> 5, c = e & 31;
    f[it] = *(const float4*)(src + (size_t)r * ld + c);
  }
#pragma unroll
  for (int it = 0; it < IT; ++it) {
    int e = (it * 128 + tid) * 4;
    int r = e >> 5, c = e & 31;
    v4bf bv;
    bv[0] = (__bf16)f[it].x; bv[1] = (__bf16)f[it].y;
    bv[2] = (__bf16)f[it].z; bv[3] = (__bf16)f[it].w;
    *(v4bf*)(dst + r * LDT + c) = bv;
  }
}

// bf16 path: async global->LDS (no VGPR round trip), tracked by ASYNCcnt.
template <int ROWS>
__device__ __forceinline__ void stage_bf16_async(const __bf16* __restrict__ src, int ld,
                                                 __bf16* dst, int tid) {
  constexpr int IT = ROWS / 32;  // per iter: 128 thr * 16B = 32 rows
#pragma unroll
  for (int it = 0; it < IT; ++it) {
    int e = (it * 128 + tid) * 8;
    int r = e >> 5, c = e & 31;
    unsigned lds = (unsigned)(size_t)(dst + r * LDT + c);
    const __bf16* g = src + (size_t)r * ld + c;
    asm volatile("global_load_async_to_lds_b128 %0, %1, off"
                 :: "v"(lds), "v"(g) : "memory");
  }
}

__device__ __forceinline__ void wait_async() {
  asm volatile("s_wait_asynccnt 0x0" ::: "memory");
}

// ---- WMMA fragment loads from LDS (per ISA 16-bit A / B layouts) -------------

__device__ __forceinline__ v16bf load_a_frag(const __bf16* lds, int lane) {
  const int row = lane & 15, kb = (lane >> 4) * 8;
  v8bf lo = *(const v8bf*)(lds + row * LDT + kb);
  v8bf hi = *(const v8bf*)(lds + row * LDT + kb + 16);
  v16bf a;
#pragma unroll
  for (int i = 0; i < 8; ++i) { a[i] = lo[i]; a[i + 8] = hi[i]; }
  return a;
}

__device__ __forceinline__ v16bf load_b_frag(const __bf16* lds, int lane) {
  const int col = lane & 15, kb = (lane >> 4) * 8;
  v8bf lo = *(const v8bf*)(lds + col * LDT + kb);
  v8bf hi = *(const v8bf*)(lds + col * LDT + kb + 16);
  v16bf b;
#pragma unroll
  for (int i = 0; i < 8; ++i) { b[i] = lo[i]; b[i + 8] = hi[i]; }
  return b;
}

// ---- per-K-step micro kernel: wave computes 32x64 (2x4 WMMA frags) -----------

__device__ __forceinline__ void mma_step(const __bf16* As, const __bf16* Bs,
                                         int wr, int wc, int lane, v8f acc[2][4]) {
  v16bf a0 = load_a_frag(As + wr * LDT, lane);
  v16bf a1 = load_a_frag(As + (wr + 16) * LDT, lane);
#pragma unroll
  for (int j = 0; j < 4; ++j) {
    v16bf bf = load_b_frag(Bs + (wc + j * 16) * LDT, lane);
    acc[0][j] = WMMA_BF16(a0, bf, acc[0][j]);
    acc[1][j] = WMMA_BF16(a1, bf, acc[1][j]);
  }
}

// ---- epilogue: 64x128 f32 acc -> bf16 row-major dst via LDS transpose --------
// Converts 64 scattered b16 stores/thread into 8 coalesced b128 stores/thread.

__device__ __forceinline__ void store_tile_via_lds(
    const v8f acc[2][4], __bf16* ts, __bf16* __restrict__ dst, int ldd,
    int row0g, int col0g, int wr, int wc, int lane, int tid) {
  const int lr = (lane >> 4) * 8, lc = lane & 15;
  __syncthreads();  // staging LDS no longer needed; safe to overwrite
#pragma unroll
  for (int i = 0; i < 2; ++i)
#pragma unroll
    for (int j = 0; j < 4; ++j) {
      __bf16* p = ts + (wr + i * 16 + lr) * LDE + (wc + j * 16 + lc);
#pragma unroll
      for (int v = 0; v < 8; ++v) p[v * LDE] = (__bf16)acc[i][j][v];
    }
  __syncthreads();
#pragma unroll
  for (int it = 0; it < 8; ++it) {
    int cidx = it * 128 + tid;           // 1024 chunks of 8 bf16
    int r = cidx >> 4, c8 = (cidx & 15) * 8;
    v8bf pk = *(const v8bf*)(ts + r * LDE + c8);
    *(v8bf*)(dst + (size_t)(row0g + r) * ldd + col0g + c8) = pk;
  }
}

// ---- kernel 1: qkv = x @ w_qkv^T + b_qkv ; Q,K row-major bf16, V transposed --

__global__ __launch_bounds__(128) void qkv_gemm_kernel(
    const float* __restrict__ X, const float* __restrict__ W, const float* __restrict__ bias,
    __bf16* __restrict__ Q, __bf16* __restrict__ Kb, __bf16* __restrict__ VT) {
  __shared__ __bf16 smem[SMEM_ELEMS];
  const int tid = threadIdx.x, lane = tid & 31, wave = tid >> 5;
  const int wr = (wave >> 1) * 32, wc = (wave & 1) * 64;
  const int row0 = blockIdx.y * 64, col0 = blockIdx.x * 128;
  v8f acc[2][4] = {};
  for (int k0 = 0, buf = 0; k0 < 1024; k0 += 32, buf ^= 1) {
    __bf16* Asb = smem + buf * ASZ;
    __bf16* Bsb = smem + BOFF + buf * BSZ;
    stage_f32<64>(X + (size_t)row0 * 1024 + k0, 1024, Asb, tid);
    stage_f32<128>(W + (size_t)col0 * 1024 + k0, 1024, Bsb, tid);
    __syncthreads();
    mma_step(Asb, Bsb, wr, wc, lane, acc);
  }
  const int lr = (lane >> 4) * 8, lc = lane & 15;
  // fold bias into the accumulators (uniform per column)
#pragma unroll
  for (int i = 0; i < 2; ++i)
#pragma unroll
    for (int j = 0; j < 4; ++j) {
      const float bv = bias[col0 + wc + j * 16 + lc];
#pragma unroll
      for (int v = 0; v < 8; ++v) acc[i][j][v] += bv;
    }
  const int which = col0 >> 10;  // uniform per block: 0=Q, 1=K, 2=V
  if (which == 2) {
    // Vt[b][c][t]: the 8 acc elements are consecutive t -> one b128 store each
    const int bb = row0 >> 11, tbase = (row0 & 2047);
#pragma unroll
    for (int i = 0; i < 2; ++i)
#pragma unroll
      for (int j = 0; j < 4; ++j) {
        const int o = (col0 & 1023) + wc + j * 16 + lc;
        v8bf pk;
#pragma unroll
        for (int v = 0; v < 8; ++v) pk[v] = (__bf16)acc[i][j][v];
        *(v8bf*)(VT + ((size_t)bb * 1024 + o) * 2048 + tbase + wr + i * 16 + lr) = pk;
      }
  } else {
    store_tile_via_lds(acc, smem, which == 0 ? Q : Kb, 1024,
                       row0, col0 & 1023, wr, wc, lane, tid);
  }
}

// ---- kernel 2: S[b,t,s] = (q . k) / 32 ; blocks above diagonal skipped ------

__global__ __launch_bounds__(128) void scores_kernel(
    const __bf16* __restrict__ Q, const __bf16* __restrict__ Kp, float* __restrict__ S) {
  const int tr0 = blockIdx.y * 64, sc0 = blockIdx.x * 128;
  if (sc0 >= tr0 + 64) return;  // fully causally masked tile
  __shared__ __bf16 smem[SMEM_ELEMS];
  const int b = blockIdx.z;
  const int tid = threadIdx.x, lane = tid & 31, wave = tid >> 5;
  const int wr = (wave >> 1) * 32, wc = (wave & 1) * 64;
  const __bf16* Qb = Q + (size_t)b * 2048 * 1024;
  const __bf16* Kb = Kp + (size_t)b * 2048 * 1024;
  v8f acc[2][4] = {};
  for (int k0 = 0, buf = 0; k0 < 1024; k0 += 32, buf ^= 1) {
    __bf16* Asb = smem + buf * ASZ;
    __bf16* Bsb = smem + BOFF + buf * BSZ;
    stage_bf16_async<64>(Qb + (size_t)tr0 * 1024 + k0, 1024, Asb, tid);
    stage_bf16_async<128>(Kb + (size_t)sc0 * 1024 + k0, 1024, Bsb, tid);
    wait_async();
    __syncthreads();
    mma_step(Asb, Bsb, wr, wc, lane, acc);
  }
  float* Sb = S + (size_t)b * 2048 * 2048;
  const int lr = (lane >> 4) * 8, lc = lane & 15;
#pragma unroll
  for (int i = 0; i < 2; ++i)
#pragma unroll
    for (int j = 0; j < 4; ++j) {
      float* p = Sb + (size_t)(tr0 + wr + i * 16 + lr) * 2048 + sc0 + wc + j * 16 + lc;
#pragma unroll
      for (int v = 0; v < 8; ++v)
        p[(size_t)v * 2048] = acc[i][j][v] * 0.03125f;  // 1/sqrt(1024)
    }
}

// ---- kernel 3: causal row softmax in place -> E = exp(s-max), rsum = 1/sum --

__global__ __launch_bounds__(256) void softmax_kernel(float* __restrict__ S,
                                                      float* __restrict__ rsum) {
  __shared__ float red[256];
  const int row = blockIdx.x;  // b*2048 + t
  const int t = row & 2047;
  float* sr = S + (size_t)row * 2048;
  const int tid = threadIdx.x;
  float m = -1e30f;
  for (int s = tid; s <= t; s += 256) m = fmaxf(m, sr[s]);
  red[tid] = m;
  __syncthreads();
  for (int off = 128; off > 0; off >>= 1) {
    if (tid < off) red[tid] = fmaxf(red[tid], red[tid + off]);
    __syncthreads();
  }
  m = red[0];
  __syncthreads();
  float sum = 0.f;
  for (int s = tid; s < 2048; s += 256) {
    const float e = (s <= t) ? __expf(sr[s] - m) : 0.f;
    sr[s] = e;
    sum += e;
  }
  red[tid] = sum;
  __syncthreads();
  for (int off = 128; off > 0; off >>= 1) {
    if (tid < off) red[tid] += red[tid + off];
    __syncthreads();
  }
  if (tid == 0) rsum[row] = 1.f / red[0];
}

// ---- kernel 4: attn_out = (E @ V) * (1/rowsum), K-loop causally bounded -----

__global__ __launch_bounds__(128) void pv_gemm_kernel(
    const float* __restrict__ E, const __bf16* __restrict__ VT,
    const float* __restrict__ rsum, __bf16* __restrict__ O) {
  __shared__ __bf16 smem[SMEM_ELEMS];
  __shared__ float rs[64];
  const int b = blockIdx.z;
  const int row0 = blockIdx.y * 64, col0 = blockIdx.x * 128;
  const int tid = threadIdx.x, lane = tid & 31, wave = tid >> 5;
  const int wr = (wave >> 1) * 32, wc = (wave & 1) * 64;
  if (tid < 64) rs[tid] = rsum[(b << 11) + row0 + tid];  // visible after first barrier
  const float* Eb = E + (size_t)b * 2048 * 2048;
  const __bf16* Vb = VT + (size_t)b * 1024 * 2048;
  v8f acc[2][4] = {};
  const int kmax = row0 + 64;  // rows in tile only attend to s < row0+64
  for (int k0 = 0, buf = 0; k0 < kmax; k0 += 32, buf ^= 1) {
    __bf16* Asb = smem + buf * ASZ;
    __bf16* Bsb = smem + BOFF + buf * BSZ;
    stage_bf16_async<128>(Vb + (size_t)col0 * 2048 + k0, 2048, Bsb, tid);
    stage_f32<64>(Eb + (size_t)row0 * 2048 + k0, 2048, Asb, tid);
    wait_async();
    __syncthreads();
    mma_step(Asb, Bsb, wr, wc, lane, acc);
  }
  const int lr = (lane >> 4) * 8;
  // fold 1/rowsum into the accumulators (uniform per row)
#pragma unroll
  for (int i = 0; i < 2; ++i) {
    float sc[8];
#pragma unroll
    for (int v = 0; v < 8; ++v) sc[v] = rs[wr + i * 16 + lr + v];
#pragma unroll
    for (int j = 0; j < 4; ++j)
#pragma unroll
      for (int v = 0; v < 8; ++v) acc[i][j][v] *= sc[v];
  }
  store_tile_via_lds(acc, smem, O, 1024, (b << 11) + row0, col0, wr, wc, lane, tid);
}

// ---- kernel 5: out = attn_out @ w_proj^T + b_proj (f32 out) -----------------

__global__ __launch_bounds__(128) void proj_gemm_kernel(
    const __bf16* __restrict__ A, const float* __restrict__ W,
    const float* __restrict__ bias, float* __restrict__ out) {
  __shared__ __bf16 smem[SMEM_ELEMS];
  const int tid = threadIdx.x, lane = tid & 31, wave = tid >> 5;
  const int wr = (wave >> 1) * 32, wc = (wave & 1) * 64;
  const int row0 = blockIdx.y * 64, col0 = blockIdx.x * 128;
  v8f acc[2][4] = {};
  for (int k0 = 0, buf = 0; k0 < 1024; k0 += 32, buf ^= 1) {
    __bf16* Asb = smem + buf * ASZ;
    __bf16* Bsb = smem + BOFF + buf * BSZ;
    stage_bf16_async<64>(A + (size_t)row0 * 1024 + k0, 1024, Asb, tid);
    stage_f32<128>(W + (size_t)col0 * 1024 + k0, 1024, Bsb, tid);
    wait_async();
    __syncthreads();
    mma_step(Asb, Bsb, wr, wc, lane, acc);
  }
  const int lr = (lane >> 4) * 8, lc = lane & 15;
#pragma unroll
  for (int i = 0; i < 2; ++i)
#pragma unroll
    for (int j = 0; j < 4; ++j) {
      const int colg = col0 + wc + j * 16 + lc;
      const float bv = bias[colg];
      float* p = out + (size_t)(row0 + wr + i * 16 + lr) * 1024 + colg;
#pragma unroll
      for (int v = 0; v < 8; ++v) p[(size_t)v * 1024] = acc[i][j][v] + bv;
    }
}

// -----------------------------------------------------------------------------

extern "C" void kernel_launch(void* const* d_in, const int* in_sizes, int n_in,
                              void* d_out, int out_size, void* d_ws, size_t ws_size,
                              hipStream_t stream) {
  (void)in_sizes; (void)n_in; (void)out_size; (void)ws_size;
  const float* x      = (const float*)d_in[0];
  const float* w_qkv  = (const float*)d_in[1];
  const float* b_qkv  = (const float*)d_in[2];
  const float* w_proj = (const float*)d_in[3];
  const float* b_proj = (const float*)d_in[4];
  float* out = (float*)d_out;

  char* ws = (char*)d_ws;
  size_t off = 0;
  auto alloc = [&](size_t bytes) -> void* {
    void* p = ws + off;
    off = (off + bytes + 255) & ~(size_t)255;
    return p;
  };
  __bf16* Q  = (__bf16*)alloc((size_t)8192 * 1024 * 2);
  __bf16* Kb = (__bf16*)alloc((size_t)8192 * 1024 * 2);
  __bf16* VT = (__bf16*)alloc((size_t)8192 * 1024 * 2);
  float*  S  = (float*)alloc((size_t)4 * 2048 * 2048 * 4);
  __bf16* AO = (__bf16*)alloc((size_t)8192 * 1024 * 2);
  float*  RS = (float*)alloc((size_t)8192 * 4);

  qkv_gemm_kernel<<<dim3(24, 128), 128, 0, stream>>>(x, w_qkv, b_qkv, Q, Kb, VT);
  scores_kernel<<<dim3(16, 32, 4), 128, 0, stream>>>(Q, Kb, S);
  softmax_kernel<<<dim3(8192), 256, 0, stream>>>(S, RS);
  pv_gemm_kernel<<<dim3(8, 32, 4), 128, 0, stream>>>(S, VT, RS, AO);
  proj_gemm_kernel<<<dim3(8, 128), 128, 0, stream>>>(AO, w_proj, b_proj, out);
}
